// PointerGeneratorOut_65455301591515
// MI455X (gfx1250) — matrix operand order repaired and checked by reading, not compiled, not run on hardware
//
#include <hip/hip_runtime.h>
#include <hip/hip_bf16.h>
#include <math.h>

#define B_     128
#define S_     512
#define D_     1024
#define GEN_V_ 50000
#define INP_V_ 30000
#define OUT_V_ 50257
#define NTILES (GEN_V_ / 16)   // 3125 N tiles of 16

typedef __attribute__((ext_vector_type(16))) __bf16 v16bf;
typedef __attribute__((ext_vector_type(8)))  float  v8f;
typedef unsigned int u32;
typedef u32 u32x4 __attribute__((ext_vector_type(4)));
typedef int i32x8 __attribute__((ext_vector_type(8)));
typedef int i32x4 __attribute__((ext_vector_type(4)));

// ---------------- zero output ----------------
__global__ void pg_zero_out(float* __restrict__ out, long long n) {
  long long t = (long long)blockIdx.x * blockDim.x + threadIdx.x;
  if (t < n) out[t] = 0.0f;
}

// ---------------- gate: interp[b] = sigmoid(x[b,:] . W_gate + b_gate) ----------------
__global__ void pg_gate(const float* __restrict__ x, const float* __restrict__ W_gate,
                        const float* __restrict__ b_gate, float* __restrict__ interp) {
  const int b = blockIdx.x;                 // 128 blocks, 256 threads
  __shared__ float red[8];
  float s = 0.0f;
  for (int d = threadIdx.x; d < D_; d += 256) s += x[b * D_ + d] * W_gate[d];
  for (int off = 16; off > 0; off >>= 1) s += __shfl_down(s, off, 32);
  if ((threadIdx.x & 31) == 0) red[threadIdx.x >> 5] = s;
  __syncthreads();
  if (threadIdx.x == 0) {
    float tot = 0.0f;
    for (int w = 0; w < 8; ++w) tot += red[w];
    tot += b_gate[0];
    interp[b] = 1.0f / (1.0f + __expf(-tot));
  }
}

// ---------------- pack x (fp32) -> bf16 in WMMA 16-bit A-operand layout ----------------
// Flat element index t = (((ks*8 + mt)*32 + lane)*16 + j)  (per-K-step data contiguous: 8 KB per ks)
//   m = mt*16 + (lane & 15)
//   k = ks*32 + ((lane>>4)&1)*16 + j
__global__ void pg_pack_a(const float* __restrict__ x, __bf16* __restrict__ Apack) {
  const int t = blockIdx.x * 256 + threadIdx.x;       // 0 .. 131071
  if (t >= 32 * 8 * 32 * 16) return;
  const int j    = t & 15;
  const int lane = (t >> 4) & 31;
  const int mt   = (t >> 9) & 7;
  const int ks   = (t >> 12) & 31;
  const int m = mt * 16 + (lane & 15);
  const int k = ks * 32 + ((lane >> 4) & 1) * 16 + j;
  Apack[t] = (__bf16)x[m * D_ + k];
}

// ---------------- TDM: copy one K-step of packed A (8 KB, contiguous) into LDS ----------------
// D# per CDNA5 ISA:
//   group0: [1:0]=count=1, [63:32]=lds_addr(bytes), [120:64]=global_addr, [127:126]=type=2
//   group1: data_size=4B (code 2); tensor_dim0=2048, tile_dim0=2048, tile_dim1=1, dim0_stride=2048
__device__ __forceinline__ void pg_tdm_load_a(const void* gsrc, unsigned lds_byte_off) {
  const unsigned long long ga = (unsigned long long)(uintptr_t)gsrc;
  u32x4 g0;
  g0[0] = 1u;                                               // count=1, user mode
  g0[1] = lds_byte_off;                                     // LDS destination (bytes)
  g0[2] = (u32)ga;                                          // global_addr[31:0]
  g0[3] = (u32)((ga >> 32) & 0x01ffffffu) | (2u << 30);     // global_addr[56:32] | type=2
  i32x8 g1;
  g1[0] = (int)(2u << 16);            // workgroup_mask=0, data_size=2 (4 bytes), no flags
  g1[1] = (int)(2048u << 16);         // [31:16] = tensor_dim0[15:0] = 2048
  g1[2] = (int)(1u << 16);            // [15:0] = tensor_dim0[31:16]=0 ; [31:16] = tensor_dim1[15:0]=1
  g1[3] = (int)(2048u << 16);         // [15:0] = tensor_dim1[31:16]=0 ; [31:16] = tile_dim0 = 2048
  g1[4] = (int)(1u);                  // [15:0] = tile_dim1 = 1 ; [31:16] = tile_dim2 = 0
  g1[5] = (int)(2048u);               // tensor_dim0_stride[31:0] = 2048 (elements)
  g1[6] = 0;                          // stride hi / tensor_dim1_stride lo
  g1[7] = 0;
  const i32x4 z4 = {0, 0, 0, 0};      // groups 2/3 unused (<=2D tensor)
  const i32x8 z8 = {0, 0, 0, 0, 0, 0, 0, 0};
  __builtin_amdgcn_tensor_load_to_lds(g0, g1, z4, z4, z8, 0);
}

// ---------------- main GEMM: logits = x @ W_gen + b_gen via v_wmma_f32_16x16x32_bf16 ----------------
// 8 waves/block, each wave one 16-wide N strip; A staged per-K-step into double-buffered LDS by TDM.
__global__ void __launch_bounds__(256) pg_gemm_wmma(const float* __restrict__ W,      // [D_, GEN_V_]
                                                    const __bf16* __restrict__ Apack, // packed A
                                                    const float* __restrict__ b_gen,  // [GEN_V_]
                                                    float* __restrict__ logits) {     // [B_, GEN_V_]
  __shared__ __align__(32) unsigned char Abuf[2][8192];
  const int lane  = threadIdx.x & 31;
  const int wid   = threadIdx.x >> 5;
  const int nt    = blockIdx.x * 8 + wid;
  const bool active = (nt < NTILES);          // wave-uniform; inactive waves still join barriers
  const int n0    = active ? nt * 16 : 0;
  const int ncol  = lane & 15;
  const int khalf = lane >> 4;
  const unsigned lds0 = (unsigned)(uintptr_t)(void*)&Abuf[0][0];  // addr[31:0] = LDS byte offset
  const unsigned lds1 = (unsigned)(uintptr_t)(void*)&Abuf[1][0];

  v8f acc[8];
#pragma unroll
  for (int mt = 0; mt < 8; ++mt)
#pragma unroll
    for (int r = 0; r < 8; ++r) acc[mt][r] = 0.0f;

  // Preload K-step 0 into buffer 0
  if (wid == 0) {
    pg_tdm_load_a((const unsigned char*)Apack, lds0);
    __builtin_amdgcn_s_wait_tensorcnt(0);
  }
  __syncthreads();

  for (int ks = 0; ks < 32; ++ks) {
    // Kick off the TDM for the next K-step; it runs during this iteration's compute.
    if (wid == 0 && ks < 31)
      pg_tdm_load_a((const unsigned char*)Apack + (size_t)(ks + 1) * 8192,
                    ((ks + 1) & 1) ? lds1 : lds0);

    if (active) {
      // B operand (32x16 bf16): element j of lane l is W[k][n], k = ks*32 + 16*khalf + j, n = n0+ncol
      const float* wp = W + (size_t)(ks * 32 + khalf * 16) * GEN_V_ + (n0 + ncol);
      v16bf bmat;
#pragma unroll
      for (int j = 0; j < 16; ++j) bmat[j] = (__bf16)wp[(size_t)j * GEN_V_];
      if (ks < 31) __builtin_prefetch(wp + (size_t)32 * GEN_V_, 0, 1);  // global_prefetch_b8

      // Hoist all 8 A tiles from LDS so the WMMAs issue back-to-back.
      const unsigned char* abase = &Abuf[ks & 1][lane * 32];
      v16bf amat[8];
#pragma unroll
      for (int mt = 0; mt < 8; ++mt)
        amat[mt] = *(const v16bf*)(abase + mt * 1024);

#pragma unroll
      for (int mt = 0; mt < 8; ++mt)
        acc[mt] = __builtin_amdgcn_wmma_f32_16x16x32_bf16(
            /*neg_a=*/false, amat[mt], /*neg_b=*/false, bmat,
            /*c_mod=*/(short)0, acc[mt], /*reuse_a=*/false, /*reuse_b=*/false);
    }

    // Make the next buffer visible to all waves before anyone reads it.
    if (wid == 0 && ks < 31) __builtin_amdgcn_s_wait_tensorcnt(0);
    __syncthreads();
  }

  if (active) {
    const float bg = b_gen[n0 + ncol];
    // C/D layout: element r of lane l is (m = mt*16 + r + 8*khalf, n = n0 + ncol)
#pragma unroll
    for (int mt = 0; mt < 8; ++mt) {
#pragma unroll
      for (int r = 0; r < 8; ++r) {
        const int m = mt * 16 + r + 8 * khalf;
        logits[(size_t)m * GEN_V_ + n0 + ncol] = acc[mt][r] + bg;
      }
    }
  }
}

// ---------------- per-row softmax stats: rowmax, rowsum ----------------
__global__ void pg_softmax_reduce(const float* __restrict__ logits,
                                  float* __restrict__ rowmax, float* __restrict__ rowsum) {
  const int b = blockIdx.x;                 // 128 blocks, 256 threads
  const float* row = logits + (size_t)b * GEN_V_;
  __shared__ float red[8];
  __shared__ float bcast;

  float mx = -3.4e38f;
  for (int v = threadIdx.x; v < GEN_V_; v += 256) mx = fmaxf(mx, row[v]);
  for (int off = 16; off > 0; off >>= 1) mx = fmaxf(mx, __shfl_down(mx, off, 32));
  if ((threadIdx.x & 31) == 0) red[threadIdx.x >> 5] = mx;
  __syncthreads();
  if (threadIdx.x == 0) {
    float m = red[0];
    for (int w = 1; w < 8; ++w) m = fmaxf(m, red[w]);
    bcast = m;
  }
  __syncthreads();
  const float m = bcast;
  __syncthreads();

  float s = 0.0f;
  for (int v = threadIdx.x; v < GEN_V_; v += 256) s += __expf(row[v] - m);
  for (int off = 16; off > 0; off >>= 1) s += __shfl_down(s, off, 32);
  if ((threadIdx.x & 31) == 0) red[threadIdx.x >> 5] = s;
  __syncthreads();
  if (threadIdx.x == 0) {
    float tot = 0.0f;
    for (int w = 0; w < 8; ++w) tot += red[w];
    rowmax[b] = m;
    rowsum[b] = tot;
  }
}

// ---------------- scatter gen probs: out[b, g2o[v]] += interp[b] * softmax(logits)[b,v] ----------------
__global__ void pg_scatter_gen(const float* __restrict__ logits, const int* __restrict__ g2o,
                               const float* __restrict__ rowmax, const float* __restrict__ rowsum,
                               const float* __restrict__ interp, float* __restrict__ out) {
  const int v = blockIdx.x * 256 + threadIdx.x;
  const int b = blockIdx.y;
  if (v >= GEN_V_) return;
  const float scale = interp[b] / rowsum[b];
  const float p = __expf(logits[(size_t)b * GEN_V_ + v] - rowmax[b]) * scale;
  atomicAdd(&out[(size_t)b * OUT_V_ + g2o[v]], p);
}

// ---------------- scatter pointer probs: out[b, i2o[ctx[b,s]]] += (1-interp[b]) * alphas[b,s] ----------------
__global__ void pg_scatter_ptr(const float* __restrict__ alphas, const int* __restrict__ ctx_inp,
                               const int* __restrict__ i2o, const float* __restrict__ interp,
                               float* __restrict__ out) {
  const int t = blockIdx.x * 256 + threadIdx.x;    // B_*S_ = 65536
  if (t >= B_ * S_) return;
  const int b = t >> 9;          // S_ = 512
  const int s = t & (S_ - 1);
  const float w = (1.0f - interp[b]) * alphas[b * S_ + s];
  const int idx = i2o[ctx_inp[b * S_ + s]];
  atomicAdd(&out[(size_t)b * OUT_V_ + idx], w);
}

// ---------------- launch ----------------
extern "C" void kernel_launch(void* const* d_in, const int* in_sizes, int n_in,
                              void* d_out, int out_size, void* d_ws, size_t ws_size,
                              hipStream_t stream) {
  const float* x       = (const float*)d_in[0];
  const float* alphas  = (const float*)d_in[1];
  const float* W_gate  = (const float*)d_in[2];
  const float* b_gate  = (const float*)d_in[3];
  const float* W_gen   = (const float*)d_in[4];
  const float* b_gen   = (const float*)d_in[5];
  const int*   ctx_inp = (const int*)d_in[6];
  const int*   g2o     = (const int*)d_in[7];
  const int*   i2o     = (const int*)d_in[8];
  float* out = (float*)d_out;

  char* ws = (char*)d_ws;
  float*  interp = (float*)(ws + 0);             // 128 f
  float*  rowmax = (float*)(ws + 512);           // 128 f
  float*  rowsum = (float*)(ws + 1024);          // 128 f
  __bf16* Apack  = (__bf16*)(ws + 2048);         // 131072 bf16 = 256 KB
  float*  logits = (float*)(ws + 2048 + 262144); // 128*50000 f = 25.6 MB

  const long long out_n = (long long)B_ * OUT_V_;
  pg_zero_out<<<(int)((out_n + 255) / 256), 256, 0, stream>>>(out, out_n);
  pg_gate<<<B_, 256, 0, stream>>>(x, W_gate, b_gate, interp);
  pg_pack_a<<<(32 * 8 * 32 * 16) / 256, 256, 0, stream>>>(x, Apack);
  pg_gemm_wmma<<<(NTILES + 7) / 8, 256, 0, stream>>>(W_gen, Apack, b_gen, logits);
  pg_softmax_reduce<<<B_, 256, 0, stream>>>(logits, rowmax, rowsum);
  pg_scatter_gen<<<dim3((GEN_V_ + 255) / 256, B_), 256, 0, stream>>>(logits, g2o, rowmax, rowsum, interp, out);
  pg_scatter_ptr<<<(B_ * S_) / 256, 256, 0, stream>>>(alphas, ctx_inp, i2o, interp, out);
}
